// MHA_16930761081343
// MI455X (gfx1250) — compile-verified, matrix operand order checked by reading, not run on hardware
//
#include <hip/hip_runtime.h>
#include <hip/hip_bf16.h>

// MHA: B=4, S=2048, H=12, D=64. Flash-attention with bf16 WMMA, fp32 accum.
// K/V tiles staged through LDS with CDNA5 async global->LDS loads (ASYNCcnt),
// double-buffered across the 32-key loop.
//
// Workspace (d_ws): q_bf16 [B*H,S,D], k_bf16 [B*H,S,D], vT_bf16 [B*H,D,S],
// then bf16 copies of Wq/Wk/Wv -> ~38 MB total.

#define B_ 4
#define S_ 2048
#define H_ 12
#define D_ 64
#define TD_ (H_ * D_)  // 768

typedef __attribute__((ext_vector_type(16))) __bf16    v16bf;
typedef __attribute__((ext_vector_type(8)))  float     v8f;
typedef __attribute__((ext_vector_type(4)))  unsigned  v4u;

union Frag {
    v16bf  v;
    v4u    q[2];
    __bf16 h[16];
};

__device__ __forceinline__ unsigned short bf_bits(float f) {
    union { __bf16 b; unsigned short u; } c;
    c.b = (__bf16)f;
    return c.u;
}

__device__ __forceinline__ void cvt8(const float* __restrict__ p, __bf16* dst) {
    float4 a = *(const float4*)p;
    float4 b = *(const float4*)(p + 4);
    dst[0] = (__bf16)a.x; dst[1] = (__bf16)a.y;
    dst[2] = (__bf16)a.z; dst[3] = (__bf16)a.w;
    dst[4] = (__bf16)b.x; dst[5] = (__bf16)b.y;
    dst[6] = (__bf16)b.z; dst[7] = (__bf16)b.w;
}

__device__ __forceinline__ v8f wmma_bf16(const v16bf& a, const v16bf& b, v8f c) {
    return __builtin_amdgcn_wmma_f32_16x16x32_bf16(false, a, false, b, (short)0, c,
                                                   false, false);
}

// async global -> LDS copy of one 16-byte chunk (per lane), tracked by ASYNCcnt
__device__ __forceinline__ void async_b128(const unsigned short* gptr,
                                           unsigned short* lptr) {
    unsigned lds = (unsigned)(unsigned long long)lptr;  // addr[31:0] = LDS offset
    asm volatile("global_load_async_to_lds_b128 %0, %1, off"
                 :: "v"(lds), "v"(gptr) : "memory");
}
__device__ __forceinline__ void wait_async0() {
    asm volatile("s_wait_asynccnt 0x0" ::: "memory");
}

// ---------------------------------------------------------------------------
// Kernel 0: one-time fp32 -> bf16 conversion of the weight matrices.
// grid (48, 3): 48*256 threads * 4 elems = 49152 = H*D*D per tensor.
// ---------------------------------------------------------------------------
__global__ __launch_bounds__(256) void wcvt_kernel(
    const float* __restrict__ Wq, const float* __restrict__ Wk,
    const float* __restrict__ Wv,
    unsigned short* __restrict__ wqb, unsigned short* __restrict__ wkb,
    unsigned short* __restrict__ wvb)
{
    const int i = (blockIdx.x * 256 + threadIdx.x) * 4;
    const float* src;
    unsigned short* dst;
    if (blockIdx.y == 0)      { src = Wq; dst = wqb; }
    else if (blockIdx.y == 1) { src = Wk; dst = wkb; }
    else                      { src = Wv; dst = wvb; }
    float4 v = *(const float4*)(src + i);
    dst[i + 0] = bf_bits(v.x);
    dst[i + 1] = bf_bits(v.y);
    dst[i + 2] = bf_bits(v.z);
    dst[i + 3] = bf_bits(v.w);
}

// ---------------------------------------------------------------------------
// One projection: out[s][e] = sum_d X[s][d] * W[e][d] + bias[e]
// TRANSPOSE=false -> [S, D]; TRANSPOSE=true -> [D, S] (for V)
// ---------------------------------------------------------------------------
template <bool TRANSPOSE>
__device__ __forceinline__ void proj_one(
    const v16bf xa0, const v16bf xa1,
    const unsigned short* __restrict__ W,  // [D, D] bf16 for this head
    const float* __restrict__ bias,        // [D] f32 for this head
    unsigned short* __restrict__ out,      // base for this (b,h)
    int st, int lm, int half, int d0)
{
    v8f acc[4];
    #pragma unroll
    for (int t = 0; t < 4; ++t) acc[t] = {};

    #pragma unroll
    for (int ks = 0; ks < 2; ++ks) {
        const v16bf& xa = ks ? xa1 : xa0;
        #pragma unroll
        for (int t = 0; t < 4; ++t) {
            // B fragment: B[k=d][n=e] = W[e][d]; lane holds column e = t*16+lm
            const unsigned short* wrow = W + ((size_t)t * 16 + lm) * D_ + ks * 32;
            Frag wb;
            wb.q[0] = *(const v4u*)(wrow + d0);
            wb.q[1] = *(const v4u*)(wrow + 16 + d0);
            acc[t] = wmma_bf16(xa, wb.v, acc[t]);
        }
    }

    #pragma unroll
    for (int t = 0; t < 4; ++t) {
        const float bv = bias[t * 16 + lm];
        #pragma unroll
        for (int vi = 0; vi < 8; ++vi) {
            const float val = acc[t][vi] + bv;
            const int s = st * 16 + vi + 8 * half;
            const int e = t * 16 + lm;
            if constexpr (!TRANSPOSE)
                out[(size_t)s * D_ + e] = bf_bits(val);
            else
                out[(size_t)e * S_ + s] = bf_bits(val);
        }
    }
}

// ---------------------------------------------------------------------------
// Kernel 1: per-head QKV projection. One wave per (b, s-tile, h).
// ---------------------------------------------------------------------------
__global__ __launch_bounds__(256) void qkv_proj_kernel(
    const float* __restrict__ x,                                 // [B, S, TD]
    const unsigned short* __restrict__ wqb, const float* __restrict__ bq,
    const unsigned short* __restrict__ wkb, const float* __restrict__ bk,
    const unsigned short* __restrict__ wvb, const float* __restrict__ bv,
    unsigned short* __restrict__ qo,    // [B*H, S, D] bf16
    unsigned short* __restrict__ ko,    // [B*H, S, D] bf16
    unsigned short* __restrict__ vto)   // [B*H, D, S] bf16
{
    const int lane = threadIdx.x & 31;
    const int wave = threadIdx.x >> 5;
    const int half = lane >> 4;
    const int lm   = lane & 15;
    const int d0   = half * 8;

    const int tile = blockIdx.x * 8 + wave;    // 0 .. 6143, h fastest
    const int h  = tile % H_;
    const int t2 = tile / H_;
    const int st = t2 % (S_ / 16);
    const int b  = t2 / (S_ / 16);
    const int bh = b * H_ + h;

    // A fragments of X (16 rows x 64 d, two K=32 fragments)
    const float* xrow = x + ((size_t)b * S_ + (size_t)st * 16 + lm) * TD_ + h * D_;
    Frag xa0, xa1;
    cvt8(xrow + d0,      xa0.h);
    cvt8(xrow + 16 + d0, xa0.h + 8);
    cvt8(xrow + 32 + d0, xa1.h);
    cvt8(xrow + 48 + d0, xa1.h + 8);

    const size_t wOff = (size_t)h * D_ * D_;
    const size_t bOff = (size_t)h * D_;
    proj_one<false>(xa0.v, xa1.v, wqb + wOff, bq + bOff,
                    qo + (size_t)bh * S_ * D_, st, lm, half, d0);
    proj_one<false>(xa0.v, xa1.v, wkb + wOff, bk + bOff,
                    ko + (size_t)bh * S_ * D_, st, lm, half, d0);
    proj_one<true >(xa0.v, xa1.v, wvb + wOff, bv + bOff,
                    vto + (size_t)bh * D_ * S_, st, lm, half, d0);
}

// ---------------------------------------------------------------------------
// Kernel 2: flash attention. 8 waves/block share one (b,h); each wave owns a
// 16-row query tile. K/V tiles (32 keys) staged into LDS by async loads,
// double-buffered; online softmax; P re-layout via LDS.
// ---------------------------------------------------------------------------
__global__ __launch_bounds__(256) void flash_attn_kernel(
    const unsigned short* __restrict__ qo,   // [B*H, S, D] bf16
    const unsigned short* __restrict__ ko,   // [B*H, S, D] bf16
    const unsigned short* __restrict__ vto,  // [B*H, D, S] bf16
    float* __restrict__ out)                 // [B, S, TD] f32
{
    // padded rows: 16B-aligned starts, conflict-free b128 reads
    __shared__ __align__(16) unsigned short Kbuf[2][32][72]; // 32 keys x 64 d
    __shared__ __align__(16) unsigned short Vbuf[2][64][40]; // 64 d x 32 keys
    __shared__ __align__(16) unsigned short Plds[8][16][40]; // per-wave P tile

    const int tid  = threadIdx.x;
    const int lane = tid & 31;
    const int wave = tid >> 5;
    const int half = lane >> 4;
    const int lm   = lane & 15;
    const int d0   = half * 8;

    const int bh = blockIdx.x / (S_ / 128);
    const int qb = blockIdx.x % (S_ / 128);
    const int q0 = (qb * 8 + wave) * 16;

    const unsigned short* qbase = qo  + (size_t)bh * S_ * D_;
    const unsigned short* kbase = ko  + (size_t)bh * S_ * D_;
    const unsigned short* vbase = vto + (size_t)bh * D_ * S_;

    // cooperative async stage of one 32-key tile (K: 256 chunks, V: 256 chunks)
    const int krow = tid >> 3, kch = (tid & 7) * 8;   // K: row 0..31, 8 chunks
    const int vrow = tid >> 2, vch = (tid & 3) * 8;   // V: row d 0..63, 4 chunks
    auto stage_tile = [&](int j, int buf) {
        async_b128(kbase + (size_t)(j + krow) * D_ + kch, &Kbuf[buf][krow][kch]);
        async_b128(vbase + (size_t)vrow * S_ + j + vch,   &Vbuf[buf][vrow][vch]);
    };

    // Q A-fragments (row m = lm, d split 0..31 / 32..63)
    Frag qa0, qa1;
    {
        const unsigned short* qrow = qbase + (size_t)(q0 + lm) * D_;
        qa0.q[0] = *(const v4u*)(qrow + d0);
        qa0.q[1] = *(const v4u*)(qrow + 16 + d0);
        qa1.q[0] = *(const v4u*)(qrow + 32 + d0);
        qa1.q[1] = *(const v4u*)(qrow + 48 + d0);
    }

    float mrow[8], lrow[8];
    v8f acc[4];
    #pragma unroll
    for (int vi = 0; vi < 8; ++vi) { mrow[vi] = -__builtin_inff(); lrow[vi] = 0.f; }
    #pragma unroll
    for (int t = 0; t < 4; ++t) acc[t] = {};

    const float sc = 0.125f * 1.44269504088896f;  // 1/sqrt(64) * log2(e)

    stage_tile(0, 0);
    wait_async0();
    __syncthreads();

    int buf = 0;
    for (int j = 0; j < S_; j += 32) {
        if (j + 32 < S_) stage_tile(j + 32, buf ^ 1);  // overlap next tile

        // ---- S = Q K^T for 32 keys from LDS
        const unsigned short* kr0 = &Kbuf[buf][lm][0];
        const unsigned short* kr1 = &Kbuf[buf][16 + lm][0];
        Frag kb0a, kb0b, kb1a, kb1b;
        kb0a.q[0] = *(const v4u*)(kr0 + d0);       kb0a.q[1] = *(const v4u*)(kr0 + 16 + d0);
        kb0b.q[0] = *(const v4u*)(kr0 + 32 + d0);  kb0b.q[1] = *(const v4u*)(kr0 + 48 + d0);
        kb1a.q[0] = *(const v4u*)(kr1 + d0);       kb1a.q[1] = *(const v4u*)(kr1 + 16 + d0);
        kb1b.q[0] = *(const v4u*)(kr1 + 32 + d0);  kb1b.q[1] = *(const v4u*)(kr1 + 48 + d0);

        v8f c0 = {}, c1 = {};
        c0 = wmma_bf16(qa0.v, kb0a.v, c0);
        c0 = wmma_bf16(qa1.v, kb0b.v, c0);
        c1 = wmma_bf16(qa0.v, kb1a.v, c1);
        c1 = wmma_bf16(qa1.v, kb1b.v, c1);

        // ---- online softmax (row r = vi + 8*half lives across 16 lanes)
        float alpha[8];
        #pragma unroll
        for (int vi = 0; vi < 8; ++vi) {
            float a = c0[vi] * sc;
            float b = c1[vi] * sc;
            float mx = fmaxf(a, b);
            #pragma unroll
            for (int off = 1; off < 16; off <<= 1)
                mx = fmaxf(mx, __shfl_xor(mx, off, 32));
            const float mn = fmaxf(mrow[vi], mx);
            const float al = exp2f(mrow[vi] - mn);
            const float p0 = exp2f(a - mn);
            const float p1 = exp2f(b - mn);
            float rs = p0 + p1;
            #pragma unroll
            for (int off = 1; off < 16; off <<= 1)
                rs += __shfl_xor(rs, off, 32);
            lrow[vi]  = lrow[vi] * al + rs;
            mrow[vi]  = mn;
            alpha[vi] = al;
            c0[vi] = p0;
            c1[vi] = p1;
        }
        #pragma unroll
        for (int t = 0; t < 4; ++t)
            #pragma unroll
            for (int vi = 0; vi < 8; ++vi) acc[t][vi] *= alpha[vi];

        // ---- re-layout P (C-layout) -> A-fragment via LDS
        #pragma unroll
        for (int vi = 0; vi < 8; ++vi) {
            const int m = vi + 8 * half;
            Plds[wave][m][lm]      = bf_bits(c0[vi]);
            Plds[wave][m][16 + lm] = bf_bits(c1[vi]);
        }
        __syncthreads();
        Frag pa;
        {
            const unsigned short* prow = &Plds[wave][lm][0];
            pa.q[0] = *(const v4u*)(prow + d0);
            pa.q[1] = *(const v4u*)(prow + 16 + d0);
        }

        // ---- acc += P V from LDS (V rows are d; chunks are key offsets)
        #pragma unroll
        for (int t = 0; t < 4; ++t) {
            const unsigned short* vr = &Vbuf[buf][t * 16 + lm][0];
            Frag vb;
            vb.q[0] = *(const v4u*)(vr + d0);
            vb.q[1] = *(const v4u*)(vr + 16 + d0);
            acc[t] = wmma_bf16(pa.v, vb.v, acc[t]);
        }

        // next tile ready + everyone done reading this buffer
        wait_async0();
        __syncthreads();
        buf ^= 1;
    }

    // ---- epilogue: divide by row sums, write fp32 output
    const int b = bh / H_;
    const int h = bh % H_;
    float inv[8];
    #pragma unroll
    for (int vi = 0; vi < 8; ++vi) inv[vi] = 1.0f / lrow[vi];
    #pragma unroll
    for (int t = 0; t < 4; ++t) {
        #pragma unroll
        for (int vi = 0; vi < 8; ++vi) {
            const int srow = q0 + vi + 8 * half;
            out[((size_t)b * S_ + srow) * TD_ + h * D_ + t * 16 + lm] =
                acc[t][vi] * inv[vi];
        }
    }
}

extern "C" void kernel_launch(void* const* d_in, const int* in_sizes, int n_in,
                              void* d_out, int out_size, void* d_ws, size_t ws_size,
                              hipStream_t stream) {
    const float* x  = (const float*)d_in[0];
    const float* Wq = (const float*)d_in[1];
    const float* bq = (const float*)d_in[2];
    const float* Wk = (const float*)d_in[3];
    const float* bk = (const float*)d_in[4];
    const float* Wv = (const float*)d_in[5];
    const float* bv = (const float*)d_in[6];
    float* out = (float*)d_out;

    const size_t per  = (size_t)B_ * H_ * S_ * D_;  // 6.29M elems
    const size_t wper = (size_t)H_ * D_ * D_;       // 49152 elems
    unsigned short* qo  = (unsigned short*)d_ws;
    unsigned short* ko  = qo + per;
    unsigned short* vt  = ko + per;
    unsigned short* wqb = vt + per;
    unsigned short* wkb = wqb + wper;
    unsigned short* wvb = wkb + wper;

    wcvt_kernel<<<dim3(48, 3), dim3(256), 0, stream>>>(Wq, Wk, Wv, wqb, wkb, wvb);

    qkv_proj_kernel<<<dim3(768), dim3(256), 0, stream>>>(
        x, wqb, bq, wkb, bk, wvb, bv, qo, ko, vt);

    flash_attn_kernel<<<dim3(768), dim3(256), 0, stream>>>(qo, ko, vt, out);
}
